// ElmanRNN_22780506538679
// MI455X (gfx1250) — matrix operand order repaired and checked by reading, not compiled, not run on hardware
//
#include <hip/hip_runtime.h>
#include <hip/hip_bf16.h>

typedef __attribute__((ext_vector_type(16))) _Float16 v16h;
typedef __attribute__((ext_vector_type(8)))  float    v8f;
typedef __attribute__((ext_vector_type(4)))  float    f32x4;
typedef unsigned int       u32;
typedef unsigned long long u64;
typedef __attribute__((ext_vector_type(4))) u32 u32x4;
typedef __attribute__((ext_vector_type(8))) u32 u32x8;

#define B_  128
#define T_  1024
#define I_  512
#define H_  1024
#define O_  512
#define WST 1032   // LDS row stride (halves) for the persistent W_hh slice (1024+8)
#define AST 1040   // LDS row stride (halves) for the TDM-padded h slice (1024 + 2x8 pad)

__device__ __forceinline__ v8f wmma16x16x32(v16h a, v16h b, v8f c) {
  // (neg_a, A, neg_b, B, c_mod, C, reuse_a, reuse_b)
  return __builtin_amdgcn_wmma_f32_16x16x32_f16(false, a, false, b, (short)0, c, false, false);
}

union FragH { v16h v; f32x4 q[2]; };

// A fragment, 16x32 f16 tile, row-major in LDS with row stride `ldh` halves.
// ISA layout: lanes 0-15 hold row M=lane, K {0..7,16..23}; lanes 16-31 row M=lane-16, K {8..15,24..31}.
__device__ __forceinline__ v16h load_frag_a(const _Float16* base, int ldh, int lane) {
  const int r  = lane & 15;
  const int ko = (lane >> 4) << 3;            // 0 or 8
  const _Float16* p = base + r * ldh + ko;
  FragH u;
  u.q[0] = *(const f32x4*)(p);
  u.q[1] = *(const f32x4*)(p + 16);
  return u.v;
}

// B fragment, 32(K)x16(N): W stored [n][k] row-major. Lanes 0-15: col n=lane, K {0..7,16..23};
// lanes 16-31: col n=lane-16, K {8..15,24..31}.
__device__ __forceinline__ v16h load_frag_b(const _Float16* base, int ldh, int lane) {
  const int c  = lane & 15;
  const int ko = (lane >> 4) << 3;            // 0 or 8
  const _Float16* p = base + c * ldh + ko;
  FragH u;
  u.q[0] = *(const f32x4*)(p);
  u.q[1] = *(const f32x4*)(p + 16);
  return u.v;
}

// ---------------------------------------------------------------------------
// Tensor Data Mover: DMA a 2D f16 tile (width elems x height rows, given row
// stride in elements) from global memory into LDS. Inline asm form per the
// bridge doc (portable across toolchains). D# group0/group1 built per ISA
// CDNA5 ISA; groups 2/3 omitted (<=2D tensor).
// flags1 = group1 dword0: wg_mask[15:0]=0 | data_size=1(2B)<<16 | pad config.
// ---------------------------------------------------------------------------
__device__ __forceinline__ void tdm_load_2d(const void* gsrc, u32 lds_byte,
                                            u32 width_elems, u32 height_rows,
                                            u64 row_stride_elems, u32 flags1) {
  const u64 ga = (u64)(uintptr_t)gsrc;
  u32x4 g0 = { 1u,                                   // count=1, user descriptor
               lds_byte,                             // lds_addr (bytes)
               (u32)ga,                              // global_addr[31:0]
               (u32)((ga >> 32) & 0x1FFFFFFu) | 0x80000000u }; // addr[56:32] | type=2
  u32x8 g1;
  g1[0] = flags1;
  g1[1] = (width_elems & 0xFFFFu) << 16;             // tensor_dim0[15:0] (abar addr = 0)
  g1[2] = (width_elems >> 16) | ((height_rows & 0xFFFFu) << 16); // tdim0 hi | tdim1 lo
  g1[3] = (height_rows >> 16) | (width_elems << 16); // tdim1 hi | tile_dim0 = width
  g1[4] = height_rows & 0xFFFFu;                     // tile_dim1 = height, tile_dim2 = 0
  g1[5] = (u32)row_stride_elems;                     // tensor_dim0_stride[31:0]
  g1[6] = (u32)(row_stride_elems >> 32) & 0xFFFFu;   // stride0[47:32] | stride1 lo = 0
  g1[7] = 0u;
  asm volatile("tensor_load_to_lds %0, %1" :: "s"(g0), "s"(g1) : "memory");
}

#define TDM_FLAGS_PLAIN  0x00010000u   // data_size=2B, no padding
// data_size=2B | pad_enable | pad_interval=7 (256 dwords) | pad_amount=3 (4 dwords = 16B)
#define TDM_FLAGS_PAD16  (0x00010000u | (1u << 20) | (7u << 22) | (3u << 25))

__device__ __forceinline__ u32 lds_addr_of(const void* p) {
  return (u32)(uintptr_t)p;            // generic LDS address: low 32 bits = LDS offset
}

// ---------------------------------------------------------------------------
// Generic WMMA GEMM: C(MxN) = A(MxK) * Bw(NxK)^T + bias.  Block tile 128x128,
// 8 waves in a 4(M)x2(N) grid, each wave computes 32x64 via 2x4 WMMA tiles.
// ---------------------------------------------------------------------------
template <typename AT, bool STORE_HALF>
__global__ __launch_bounds__(256) void wmma_gemm_bias_kernel(
    const AT* __restrict__ A, int lda,
    const _Float16* __restrict__ Bw, int ldb,
    const float* __restrict__ bias,
    void* __restrict__ C, int ldc, int K) {
  __shared__ __align__(16) _Float16 sA[128 * 40];
  __shared__ __align__(16) _Float16 sB[128 * 40];

  const int tid  = threadIdx.x;
  const int lane = tid & 31;
  const int wave = tid >> 5;
  const int wm   = wave & 3;
  const int wn   = wave >> 2;
  const size_t m0 = (size_t)blockIdx.y * 128;
  const size_t n0 = (size_t)blockIdx.x * 128;

  const int lrow = tid >> 1;
  const int lseg = (tid & 1) * 16;

  const v8f vzero = {};
  v8f acc[2][4];
#pragma unroll
  for (int i = 0; i < 2; ++i)
#pragma unroll
    for (int j = 0; j < 4; ++j) acc[i][j] = vzero;

  for (int kk = 0; kk < K; kk += 32) {
    if constexpr (sizeof(AT) == 4) {
      const float* ap = (const float*)A + (m0 + lrow) * (size_t)lda + kk + lseg;
      f32x4 f0 = *(const f32x4*)(ap);
      f32x4 f1 = *(const f32x4*)(ap + 4);
      f32x4 f2 = *(const f32x4*)(ap + 8);
      f32x4 f3 = *(const f32x4*)(ap + 12);
      FragH h;
#pragma unroll
      for (int i = 0; i < 4; ++i) {
        h.v[i]      = (_Float16)f0[i];
        h.v[4 + i]  = (_Float16)f1[i];
        h.v[8 + i]  = (_Float16)f2[i];
        h.v[12 + i] = (_Float16)f3[i];
      }
      *(f32x4*)&sA[lrow * 40 + lseg]     = h.q[0];
      *(f32x4*)&sA[lrow * 40 + lseg + 8] = h.q[1];
    } else {
      const _Float16* ap = (const _Float16*)A + (m0 + lrow) * (size_t)lda + kk + lseg;
      *(f32x4*)&sA[lrow * 40 + lseg]     = *(const f32x4*)(ap);
      *(f32x4*)&sA[lrow * 40 + lseg + 8] = *(const f32x4*)(ap + 8);
    }
    const _Float16* bp = Bw + (n0 + lrow) * (size_t)ldb + kk + lseg;
    *(f32x4*)&sB[lrow * 40 + lseg]     = *(const f32x4*)(bp);
    *(f32x4*)&sB[lrow * 40 + lseg + 8] = *(const f32x4*)(bp + 8);
    __syncthreads();

    v16h fa[2], fb[4];
#pragma unroll
    for (int tm = 0; tm < 2; ++tm)
      fa[tm] = load_frag_a(&sA[(wm * 32 + tm * 16) * 40], 40, lane);
#pragma unroll
    for (int tn = 0; tn < 4; ++tn)
      fb[tn] = load_frag_b(&sB[(wn * 64 + tn * 16) * 40], 40, lane);
#pragma unroll
    for (int tm = 0; tm < 2; ++tm)
#pragma unroll
      for (int tn = 0; tn < 4; ++tn)
        acc[tm][tn] = wmma16x16x32(fa[tm], fb[tn], acc[tm][tn]);
    __syncthreads();
  }

  const int cn = lane & 15;
  const int hi = lane >> 4;
#pragma unroll
  for (int tm = 0; tm < 2; ++tm)
#pragma unroll
    for (int tn = 0; tn < 4; ++tn) {
      const size_t n = n0 + wn * 64 + tn * 16 + cn;
      const float bv = bias[n];
#pragma unroll
      for (int j = 0; j < 8; ++j) {
        const size_t m = m0 + wm * 32 + tm * 16 + j + 8 * hi;
        const float v = acc[tm][tn][j] + bv;
        if constexpr (STORE_HALF)
          ((_Float16*)C)[m * (size_t)ldc + n] = (_Float16)v;
        else
          ((float*)C)[m * (size_t)ldc + n] = v;
      }
    }
}

// ---------------------------------------------------------------------------
// Grid-wide phase barrier (device scope) for the persistent scan kernel.
// ---------------------------------------------------------------------------
__device__ __forceinline__ void grid_barrier(unsigned* cnt, unsigned* sense,
                                             unsigned nblk, unsigned phase) {
  __syncthreads();
  if (threadIdx.x == 0) {
    const unsigned old =
        __hip_atomic_fetch_add(cnt, 1u, __ATOMIC_ACQ_REL, __HIP_MEMORY_SCOPE_AGENT);
    if (old == nblk - 1u) {
      __hip_atomic_store(cnt, 0u, __ATOMIC_RELAXED, __HIP_MEMORY_SCOPE_AGENT);
      __hip_atomic_store(sense, phase + 1u, __ATOMIC_RELEASE, __HIP_MEMORY_SCOPE_AGENT);
    } else {
      while (__hip_atomic_load(sense, __ATOMIC_ACQUIRE, __HIP_MEMORY_SCOPE_AGENT) < phase + 1u)
        __builtin_amdgcn_s_sleep(2);
    }
  }
  __syncthreads();
}

// ---------------------------------------------------------------------------
// Persistent scan: 32 workgroups (2 M-tiles x 16 N-tiles of 64). W_hh slice
// pinned in LDS all 1024 steps; per step the TDM DMAs the 64x1024 h slice
// (padded to a 1040-half stride) and prefetches the next xw tile while the
// current step's 64 WMMAs run; the K-loop itself is barrier-free LDS+WMMA.
// ---------------------------------------------------------------------------
__global__ __launch_bounds__(256) void rnn_scan_kernel(
    const _Float16* __restrict__ xw,     // (B,T,H) f16
    const _Float16* __restrict__ Whh,    // (H,H) f16 row-major
    const float* __restrict__ bhh,
    _Float16* __restrict__ h0buf,        // (B,H) f16, h0=0 on entry
    _Float16* __restrict__ h1buf,
    unsigned* cnt, unsigned* sense) {
  extern __shared__ __align__(16) _Float16 smem[];
  _Float16* sW  = smem;                           // 64 x WST  (129 KB)
  _Float16* sA  = smem + 64 * WST;                // 64 x AST  (130 KB, TDM pad)
  _Float16* sXW = smem + 64 * WST + 64 * AST;     // 2 x 64x64 (16 KB)

  const int tid  = threadIdx.x;
  const int lane = tid & 31;
  const int wave = tid >> 5;
  const int wm   = wave & 3;             // 4 x 16 rows = 64 M
  const int wn   = wave >> 2;            // 2 x 32 cols = 64 N
  const int m0   = (blockIdx.x & 1) * 64;
  const int n0   = (blockIdx.x >> 1) * 64;

  // Pin our W_hh slice (rows n0..n0+63, full K) in LDS for the whole scan.
  for (int c = tid; c < 64 * 128; c += 256) {     // 8-half (16B) chunks
    const int r  = c >> 7;
    const int k8 = (c & 127) << 3;
    *(f32x4*)&sW[r * WST + k8] = *(const f32x4*)&Whh[(size_t)(n0 + r) * H_ + k8];
  }

  const int cn = lane & 15;
  const int hi = lane >> 4;
  float bias_r[2];
#pragma unroll
  for (int tn = 0; tn < 2; ++tn) bias_r[tn] = bhh[n0 + wn * 32 + tn * 16 + cn];

  // Prefetch xw tile for t=0.
  if (wave == 0)
    tdm_load_2d(&xw[((size_t)m0 * T_) * H_ + n0], lds_addr_of(&sXW[0]),
                64, 64, (u64)T_ * H_, TDM_FLAGS_PLAIN);
  __syncthreads();

  const v8f vzero = {};
  for (int t = 0; t < T_; ++t) {
    const _Float16* hsrc = (t & 1) ? h1buf : h0buf;
    _Float16*       hdst = (t & 1) ? h0buf : h1buf;

    if (wave == 0) {
      // DMA this step's h slice (in-order after any outstanding xw prefetch).
      tdm_load_2d(hsrc + (size_t)m0 * H_, lds_addr_of(sA),
                  1024, 64, (u64)H_, TDM_FLAGS_PAD16);
      if (t + 1 < T_) {
        // Prefetch next step's xw tile; overlaps the whole K-loop below.
        tdm_load_2d(&xw[((size_t)m0 * T_ + (t + 1)) * H_ + n0],
                    lds_addr_of(&sXW[((t + 1) & 1) * 4096]),
                    64, 64, (u64)T_ * H_, TDM_FLAGS_PLAIN);
        __builtin_amdgcn_s_wait_tensorcnt(1);   // h(t) + xw(t) done, xw(t+1) in flight
      } else {
        __builtin_amdgcn_s_wait_tensorcnt(0);
      }
    }
    __syncthreads();

    // Barrier-free K-loop: pure ds_load + v_wmma, everything LDS-resident.
    v8f acc[2] = {vzero, vzero};
#pragma unroll
    for (int kk = 0; kk < H_; kk += 32) {
      const int apad = (kk >= 512) ? 8 : 0;       // TDM pad after 256 dwords
      v16h fa = load_frag_a(&sA[(wm * 16) * AST + kk + apad], AST, lane);
#pragma unroll
      for (int tn = 0; tn < 2; ++tn) {
        v16h fb = load_frag_b(&sW[(wn * 32 + tn * 16) * WST + kk], WST, lane);
        acc[tn] = wmma16x16x32(fa, fb, acc[tn]);
      }
    }

    // h_new = relu(acc + xw_t + b_hh); xw tile comes from the TDM LDS buffer.
    const _Float16* sx = &sXW[(t & 1) * 4096];
#pragma unroll
    for (int tn = 0; tn < 2; ++tn) {
      const int nloc = wn * 32 + tn * 16 + cn;
#pragma unroll
      for (int j = 0; j < 8; ++j) {
        const int mloc = wm * 16 + j + 8 * hi;
        float v = acc[tn][j] + (float)sx[mloc * 64 + nloc] + bias_r[tn];
        v = fmaxf(v, 0.0f);
        hdst[(size_t)(m0 + mloc) * H_ + n0 + nloc] = (_Float16)v;
      }
    }
    grid_barrier(cnt, sense, 32u, (unsigned)t);
  }
}

// ---------------------------------------------------------------------------
__global__ void init_state_kernel(unsigned* flags, _Float16* h0) {
  const size_t i = (size_t)blockIdx.x * blockDim.x + threadIdx.x;
  if (i < 2) flags[i] = 0u;
  const size_t n = (size_t)B_ * H_ / 2;     // as u32
  for (size_t k = i; k < n; k += (size_t)gridDim.x * blockDim.x)
    ((unsigned*)h0)[k] = 0u;
}

__global__ void f32_to_f16_kernel(const float* __restrict__ src,
                                  _Float16* __restrict__ dst, int n) {
  int i = blockIdx.x * blockDim.x + threadIdx.x;
  const int stride = gridDim.x * blockDim.x;
  for (; i < n; i += stride) dst[i] = (_Float16)src[i];
}

// ---------------------------------------------------------------------------
extern "C" void kernel_launch(void* const* d_in, const int* in_sizes, int n_in,
                              void* d_out, int out_size, void* d_ws, size_t ws_size,
                              hipStream_t stream) {
  (void)in_sizes; (void)n_in; (void)out_size; (void)ws_size;
  const float* x    = (const float*)d_in[0];
  const float* W_ih = (const float*)d_in[1];
  const float* b_ih = (const float*)d_in[2];
  const float* W_hh = (const float*)d_in[3];
  const float* b_hh = (const float*)d_in[4];
  const float* W_ho = (const float*)d_in[5];
  const float* b_ho = (const float*)d_in[6];

  char* ws = (char*)d_ws;
  unsigned* flags = (unsigned*)ws;
  _Float16* wih_h = (_Float16*)(ws + 256);
  _Float16* whh_h = (_Float16*)(ws + 256 + 1048576);
  _Float16* who_h = (_Float16*)(ws + 256 + 1048576 + 2097152);
  _Float16* h0    = (_Float16*)(ws + 256 + 1048576 + 2097152 + 1048576);
  _Float16* h1    = (_Float16*)((char*)h0 + (size_t)B_ * H_ * 2);
  _Float16* xw    = (_Float16*)((char*)h1 + (size_t)B_ * H_ * 2);

  hipLaunchKernelGGL(init_state_kernel, dim3(64), dim3(256), 0, stream, flags, h0);
  hipLaunchKernelGGL(f32_to_f16_kernel, dim3(512),  dim3(256), 0, stream, W_ih, wih_h, H_ * I_);
  hipLaunchKernelGGL(f32_to_f16_kernel, dim3(1024), dim3(256), 0, stream, W_hh, whh_h, H_ * H_);
  hipLaunchKernelGGL(f32_to_f16_kernel, dim3(512),  dim3(256), 0, stream, W_ho, who_h, O_ * H_);

  // Phase 1: xw = x @ W_ih^T + b_ih   (M=B*T, N=H, K=I), stored f16.
  hipLaunchKernelGGL((wmma_gemm_bias_kernel<float, true>),
                     dim3(H_ / 128, (B_ * T_) / 128), dim3(256), 0, stream,
                     x, I_, wih_h, I_, b_ih, (void*)xw, H_, I_);

  // Phase 2: persistent recurrent scan over T with grid-wide barrier + TDM staging.
  const size_t lds_scan = (size_t)(64 * WST + 64 * AST + 2 * 64 * 64) * sizeof(_Float16); // ~275 KB
  hipLaunchKernelGGL(rnn_scan_kernel, dim3(32), dim3(256), lds_scan, stream,
                     xw, whh_h, b_hh, h0, h1, flags, flags + 1);

  // Phase 3: out = h_final @ W_ho^T + b_ho  (M=B, N=O, K=H), h_final in h0.
  hipLaunchKernelGGL((wmma_gemm_bias_kernel<_Float16, false>),
                     dim3(O_ / 128, B_ / 128), dim3(256), 0, stream,
                     h0, H_, who_h, H_, b_ho, d_out, O_, H_);
}
